// MultiHeadAttention_24240795419336
// MI455X (gfx1250) — compile-verified, matrix operand order checked by reading, not run on hardware
//
#include <hip/hip_runtime.h>
#include <hip/hip_bf16.h>

// MI455X / gfx1250: wave32, WMMA 16x16x32 bf16 (f32 accum), async global->LDS.
// B=2, S=2048, D=1024, H=16, HD=64 (hardcoded from the reference).

typedef __bf16 bf16_t;
typedef __attribute__((ext_vector_type(16))) __bf16 v16bf;
typedef __attribute__((ext_vector_type(8)))  __bf16 v8bf;
typedef __attribute__((ext_vector_type(8)))  float  v8f;

#define DEV static __device__ __forceinline__

// Load a 16x32 bf16 fragment (A-layout per CDNA5 ISA 7.12.2) from a row-major
// LDS tile: lane L(0..15): M=L, kb=0; lane L+16: same M, kb=8.
// Two contiguous 16B reads per lane -> ds_load_b128.
DEV v16bf load_frag(const bf16_t* base, int ld) {
  const int lane = threadIdx.x & 31;
  const int m    = lane & 15;
  const int kb   = (lane >> 4) << 3;          // 0 or 8
  const bf16_t* row = base + m * ld;
  v8bf lo = *reinterpret_cast<const v8bf*>(row + kb);
  v8bf hi = *reinterpret_cast<const v8bf*>(row + 16 + kb);
  return __builtin_shufflevector(lo, hi, 0, 1, 2, 3, 4, 5, 6, 7,
                                          8, 9, 10, 11, 12, 13, 14, 15);
}

DEV v8f wmma_bf16(v16bf a, v16bf b, v8f c) {
  return __builtin_amdgcn_wmma_f32_16x16x32_bf16(
      /*neg_a=*/false, a, /*neg_b=*/false, b,
      /*c_mod=*/(short)0, c, /*reuse_a=*/false, /*reuse_b=*/false);
}

// Async global->LDS 16-byte copy (GLOBAL_LOAD_ASYNC_TO_LDS_B128, ASYNCcnt).
// GV mode: per-lane 64-bit global address; LDS dest = low 32 bits of the
// generic shared-pointer (aperture rule).
DEV void async_copy_b128(const bf16_t* gsrc, const bf16_t* ldst) {
  unsigned lds_off = (unsigned)(size_t)ldst;
  unsigned long long gaddr = (unsigned long long)(size_t)gsrc;
  asm volatile("global_load_async_to_lds_b128 %0, %1, off"
               :: "v"(lds_off), "v"(gaddr) : "memory");
}

DEV void async_wait0() {
  asm volatile("s_wait_asynccnt 0x0" ::: "memory");
}

// ---------------------------------------------------------------------------
// GEMM: C[M,N] = A[M,K] @ W[K,N]; A is f32 (MODE 0) or bf16 (MODE 1).
// MODE 0: output scattered to bf16 [which][B][H][S][HD] (qkv->heads).
// MODE 1: output plain f32 row-major (final projection).
// Block tile 128x128, 256 threads = 8 waves (4Mx2N), 32x64 per wave,
// Kstep 32 -> 8 WMMAs per wave per K iteration.
// ---------------------------------------------------------------------------
template <int MODE>
__global__ __launch_bounds__(256)
void gemm_bf16_wmma(const void* __restrict__ Araw,
                    const float* __restrict__ W,
                    void* __restrict__ Craw,
                    int Ncols, int Kdim) {
  __shared__ alignas(16) bf16_t As[128 * 32];  // 8 KB
  __shared__ alignas(16) bf16_t Wt[128 * 32];  // 8 KB, transposed (N-major)

  const int tid  = threadIdx.x;
  const int wave = tid >> 5;
  const int wm   = (wave & 3) * 32;            // wave row offset in block tile
  const int wn   = (wave >> 2) * 64;           // wave col offset
  const int row0 = blockIdx.y * 128;
  const int col0 = blockIdx.x * 128;

  const float*  Af = (const float*)Araw;
  const bf16_t* Ab = (const bf16_t*)Araw;

  v8f acc[2][4] = {};

  for (int k0 = 0; k0 < Kdim; k0 += 32) {
    // Stage A tile (128x32) into LDS as bf16 (converting if f32 source).
#pragma unroll
    for (int it = 0; it < 16; ++it) {
      int idx = tid + it * 256;
      int r = idx >> 5, c = idx & 31;
      if constexpr (MODE == 0)
        As[idx] = (bf16_t)Af[(size_t)(row0 + r) * Kdim + k0 + c];
      else
        As[idx] = Ab[(size_t)(row0 + r) * Kdim + k0 + c];
    }
    // Stage W tile (32x128) transposed into Wt[128][32] so B-fragments are
    // contiguous b128 reads (B column n = Wt row n).
#pragma unroll
    for (int it = 0; it < 16; ++it) {
      int idx = tid + it * 256;
      int k = idx >> 7, nn = idx & 127;
      Wt[nn * 32 + k] = (bf16_t)W[(size_t)(k0 + k) * Ncols + col0 + nn];
    }
    // Prefetch next A K-tile (global_prefetch_b8).
    if (k0 + 32 < Kdim) {
      if constexpr (MODE == 0)
        __builtin_prefetch(&Af[(size_t)(row0 + (tid >> 1)) * Kdim + k0 + 32], 0, 1);
      else
        __builtin_prefetch(&Ab[(size_t)(row0 + (tid >> 1)) * Kdim + k0 + 32], 0, 1);
    }
    __syncthreads();

    v16bf af0 = load_frag(&As[(wm + 0) * 32], 32);
    v16bf af1 = load_frag(&As[(wm + 16) * 32], 32);
#pragma unroll
    for (int j = 0; j < 4; ++j) {
      v16bf wf = load_frag(&Wt[(wn + j * 16) * 32], 32);
      acc[0][j] = wmma_bf16(af0, wf, acc[0][j]);
      acc[1][j] = wmma_bf16(af1, wf, acc[1][j]);
    }

    __syncthreads();
  }

  // Epilogue. C-layout: lane holds (M = mh + r, N = n) in acc[.][.][r].
  const int lane = tid & 31;
  const int n    = lane & 15;
  const int mh   = (lane >> 4) * 8;

#pragma unroll
  for (int i = 0; i < 2; ++i) {
#pragma unroll
    for (int j = 0; j < 4; ++j) {
#pragma unroll
      for (int r = 0; r < 8; ++r) {
        int gm = row0 + wm + i * 16 + mh + r;
        int gn = col0 + wn + j * 16 + n;
        float v = acc[i][j][r];
        if constexpr (MODE == 0) {
          int which = gn >> 10;         // 0=q 1=k 2=v
          int d     = gn & 1023;
          int h     = d >> 6;
          int hd    = d & 63;
          int b     = gm >> 11;         // row = b*2048 + s
          int s     = gm & 2047;
          ((bf16_t*)Craw)[((((size_t)which * 2 + b) * 16 + h) * 2048 + s) * 64 + hd] =
              (bf16_t)v;
        } else {
          ((float*)Craw)[(size_t)gm * 1024 + gn] = v;
        }
      }
    }
  }
}

// ---------------------------------------------------------------------------
// Flash attention: grid (S/64, B*H). 128 threads = 4 waves, 16 query rows
// per wave. Q/K tiles staged via async global->LDS (ASYNCcnt); V staged
// transposed synchronously; all matmuls via WMMA with online softmax.
// ---------------------------------------------------------------------------
__global__ __launch_bounds__(128)
void flash_attn_wmma(const bf16_t* __restrict__ qkv, bf16_t* __restrict__ Ow) {
  __shared__ alignas(16) bf16_t Qs[64 * 64];       // 8 KB
  __shared__ alignas(16) bf16_t Ks[32 * 64];       // 4 KB (key-major)
  __shared__ alignas(16) bf16_t Vt[64 * 32];       // 4 KB (hd-major = transposed)
  __shared__ alignas(16) bf16_t Ps[4][16 * 32];    // 4 KB, per-wave P scratch

  const int tid  = threadIdx.x;
  const int wave = tid >> 5;
  const int lane = tid & 31;
  const int n    = lane & 15;
  const int mh   = (lane >> 4) * 8;

  const int qb = blockIdx.x;              // 64-row query tile
  const int bh = blockIdx.y;              // b*16 + h
  const int b  = bh >> 4;
  const int h  = bh & 15;

  const size_t headOff = (size_t)bh * 2048 * 64;
  const bf16_t* Qg = qkv + headOff + (size_t)qb * 64 * 64;
  const bf16_t* Kg = qkv + (size_t)2 * 16 * 2048 * 64 + headOff;
  const bf16_t* Vg = qkv + (size_t)2 * 2 * 16 * 2048 * 64 + headOff;

  // Q tile 64x64 -> LDS via async b128 copies (512 chunks / 128 threads).
#pragma unroll
  for (int it = 0; it < 4; ++it) {
    int idx = tid + it * 128;
    async_copy_b128(Qg + idx * 8, Qs + idx * 8);
  }
  async_wait0();
  __syncthreads();

  const int q0 = wave * 16;
  v16bf qf0 = load_frag(&Qs[q0 * 64 + 0],  64);   // hd 0..31
  v16bf qf1 = load_frag(&Qs[q0 * 64 + 32], 64);   // hd 32..63

  float mrun[8], lrun[8];
  v8f oacc[4] = {};
#pragma unroll
  for (int r = 0; r < 8; ++r) { mrun[r] = -3.0e38f; lrun[r] = 0.0f; }

  const float scale = 0.125f;                 // 1/sqrt(64)
  const float log2e = 1.4426950408889634f;
  const int   qg0   = qb * 64 + q0 + mh;      // first global q row of this lane
  const int   qend  = qb * 64 + 64;           // causal limit (block-uniform)

  for (int kb = 0; kb < qend; kb += 32) {
    // Stage K (32x64) via async b128; V transposed (64x32) synchronously.
#pragma unroll
    for (int it = 0; it < 2; ++it) {
      int idx = tid + it * 128;
      async_copy_b128(Kg + (size_t)kb * 64 + idx * 8, Ks + idx * 8);
    }
#pragma unroll
    for (int it = 0; it < 16; ++it) {
      int idx = tid + it * 128;
      int kk = idx >> 6, hd = idx & 63;
      Vt[hd * 32 + kk] = Vg[(size_t)(kb + kk) * 64 + hd];
    }
    async_wait0();
    __syncthreads();

    // S = Q K^T : two 16x16 score tiles, each HD=64 as two K=32 WMMAs.
    v8f sc[2];
#pragma unroll
    for (int j = 0; j < 2; ++j) {
      v16bf kf0 = load_frag(&Ks[(j * 16) * 64 + 0],  64);
      v16bf kf1 = load_frag(&Ks[(j * 16) * 64 + 32], 64);
      v8f c = {};
      c = wmma_bf16(qf0, kf0, c);
      c = wmma_bf16(qf1, kf1, c);
      sc[j] = c;
    }

    // Scale, causal mask, row-max over the 16 lanes of each half-wave.
    float p[2][8], rm[8];
#pragma unroll
    for (int r = 0; r < 8; ++r) {
      float s0 = sc[0][r] * scale;
      float s1 = sc[1][r] * scale;
      int q = qg0 + r;
      if (kb + n      > q) s0 = -3.0e38f;
      if (kb + 16 + n > q) s1 = -3.0e38f;
      p[0][r] = s0; p[1][r] = s1;
      float mx = fmaxf(s0, s1);
      mx = fmaxf(mx, __shfl_xor(mx, 1));
      mx = fmaxf(mx, __shfl_xor(mx, 2));
      mx = fmaxf(mx, __shfl_xor(mx, 4));
      mx = fmaxf(mx, __shfl_xor(mx, 8));
      rm[r] = mx;
    }

    // Online softmax update; emit P (bf16) into per-wave LDS scratch.
#pragma unroll
    for (int r = 0; r < 8; ++r) {
      float mnew  = fmaxf(mrun[r], rm[r]);
      float alpha = exp2f((mrun[r] - mnew) * log2e);
      float p0 = exp2f((p[0][r] - mnew) * log2e);
      float p1 = exp2f((p[1][r] - mnew) * log2e);
      float rs = p0 + p1;
      rs += __shfl_xor(rs, 1);
      rs += __shfl_xor(rs, 2);
      rs += __shfl_xor(rs, 4);
      rs += __shfl_xor(rs, 8);
      lrun[r] = lrun[r] * alpha + rs;
      mrun[r] = mnew;
#pragma unroll
      for (int t = 0; t < 4; ++t) oacc[t][r] *= alpha;
      Ps[wave][(mh + r) * 32 +      n] = (bf16_t)p0;
      Ps[wave][(mh + r) * 32 + 16 + n] = (bf16_t)p1;
    }
    __syncthreads();   // P visible for A-layout reload (loop is block-uniform)

    // O += P @ V : P is 16x32 A-frag; Vt rows give contiguous B-frags.
    v16bf pf = load_frag(Ps[wave], 32);
#pragma unroll
    for (int t = 0; t < 4; ++t) {
      v16bf vf = load_frag(&Vt[(t * 16) * 32], 32);
      oacc[t] = wmma_bf16(pf, vf, oacc[t]);
    }
    __syncthreads();   // done reading Ks/Vt before next stage
  }

  // Normalize and write O as bf16 [B][S][D] for the final projection.
#pragma unroll
  for (int r = 0; r < 8; ++r) {
    float inv = 1.0f / lrun[r];
    int s = qb * 64 + q0 + mh + r;
    size_t rowbase = ((size_t)b * 2048 + s) * 1024 + (size_t)h * 64;
#pragma unroll
    for (int t = 0; t < 4; ++t)
      Ow[rowbase + t * 16 + n] = (bf16_t)(oacc[t][r] * inv);
  }
}

// ---------------------------------------------------------------------------
extern "C" void kernel_launch(void* const* d_in, const int* in_sizes, int n_in,
                              void* d_out, int out_size, void* d_ws, size_t ws_size,
                              hipStream_t stream) {
  (void)in_sizes; (void)n_in; (void)out_size; (void)ws_size;

  const float* x     = (const float*)d_in[0];   // (2, 2048, 1024) f32
  const float* w_qkv = (const float*)d_in[1];   // (1024, 3072)   f32
  const float* w_out = (const float*)d_in[2];   // (1024, 1024)   f32
  float* out = (float*)d_out;                   // (2, 2048, 1024) f32

  bf16_t* qkv_ws = (bf16_t*)d_ws;                          // 12,582,912 bf16 (~25.2 MB)
  bf16_t* o_ws   = qkv_ws + (size_t)3 * 2 * 2048 * 1024;   //  4,194,304 bf16 (~8.4 MB)

  // 1) qkv = x @ w_qkv (M=4096, N=3072, K=1024), scattered to heads.
  gemm_bf16_wmma<0><<<dim3(3072 / 128, 4096 / 128), 256, 0, stream>>>(
      x, w_qkv, qkv_ws, 3072, 1024);

  // 2) Fused causal flash attention per (b, h, 64-query tile).
  flash_attn_wmma<<<dim3(2048 / 64, 2 * 16), 128, 0, stream>>>(qkv_ws, o_ws);

  // 3) out = O @ w_out (M=4096, N=1024, K=1024), f32 result.
  gemm_bf16_wmma<1><<<dim3(1024 / 128, 4096 / 128), 256, 0, stream>>>(
      o_ws, w_out, out, 1024, 1024);
}
